// Codebook_32384053412096
// MI455X (gfx1250) — matrix operand order repaired and checked by reading, not compiled, not run on hardware
//
#include <hip/hip_runtime.h>

#define CODE_DIM 256
#define KCODES   1024
#define NROWS    32768
#define HW       1024
#define M_TILE   64
#define THREADS  128
#define OUT_ELEMS 8388608   // 32*256*32*32

typedef __attribute__((ext_vector_type(16))) __bf16       v16bf;
typedef __attribute__((ext_vector_type(8)))  float        v8f;
typedef __attribute__((ext_vector_type(4)))  unsigned int v4u;

union ABu { v16bf v; v4u q[2]; };

__device__ __forceinline__ unsigned short f2bf(float f) {
  unsigned u = __float_as_uint(f);
  unsigned r = 0x7FFFu + ((u >> 16) & 1u);
  return (unsigned short)((u + r) >> 16);
}
__device__ __forceinline__ float bf2f(unsigned short h) {
  return __uint_as_float(((unsigned)h) << 16);
}

// ---------------------------------------------------------------------------
// Prep: split weight into bf16 hi+lo, compute ||w_k||^2. One wave per row.
// ---------------------------------------------------------------------------
__global__ void prep_weight_kernel(const float* __restrict__ w,
                                   unsigned short* __restrict__ w_hi,
                                   unsigned short* __restrict__ w_lo,
                                   float* __restrict__ wnorm) {
  const int k    = blockIdx.x * 8 + (threadIdx.x >> 5);
  const int lane = threadIdx.x & 31;
  const float* row = w + k * CODE_DIM;
  float s = 0.f;
#pragma unroll
  for (int j = 0; j < 8; ++j) {
    int c = lane + 32 * j;
    float f = row[c];
    s += f * f;
    unsigned short h = f2bf(f);
    w_hi[k * CODE_DIM + c] = h;
    w_lo[k * CODE_DIM + c] = f2bf(f - bf2f(h));
  }
#pragma unroll
  for (int m = 16; m >= 1; m >>= 1) s += __shfl_xor(s, m, 32);
  if (lane == 0) wnorm[k] = s;
}

// ---------------------------------------------------------------------------
// Main: fused split-bf16 WMMA GEMM + argmin over codes.
// Block = 128 threads (4 waves), tile = 64 rows. Wave wv handles k-tiles
// wv, wv+4, ... across all 4 row-tiles (weight read once per block).
// ---------------------------------------------------------------------------
__global__ __launch_bounds__(THREADS) void argmin_kernel(
    const float* __restrict__ z,
    const unsigned short* __restrict__ w_hi,
    const unsigned short* __restrict__ w_lo,
    const float* __restrict__ wnorm,
    int* __restrict__ idx_out) {
  __shared__ unsigned short lds_hi[M_TILE * CODE_DIM];  // 32 KB
  __shared__ unsigned short lds_lo[M_TILE * CODE_DIM];  // 32 KB

  const int tid  = threadIdx.x;
  const int wv   = tid >> 5;
  const int lane = tid & 31;
  const int n0   = blockIdx.x * M_TILE;
  const int b    = n0 >> 10;        // batch image
  const int hw0  = n0 & (HW - 1);
  const float* zb = z + (size_t)b * CODE_DIM * HW + hw0;

  // Stage z tile -> LDS as split bf16, XOR-swizzled (16B blocks) so the
  // per-lane ds_load_b128 A-operand fetches are bank-conflict free.
  for (int e = tid; e < M_TILE * CODE_DIM; e += THREADS) {
    int c = e >> 6;   // channel 0..255
    int i = e & 63;   // row in tile (coalesced global reads over i)
    float f = zb[c * HW + i];
    unsigned short h = f2bf(f);
    unsigned short l = f2bf(f - bf2f(h));
    int off = i * CODE_DIM + (((c >> 3) ^ (i & 15)) << 3) + (c & 7);
    lds_hi[off] = h;
    lds_lo[off] = l;
  }
  __syncthreads();

  const int lrow  = lane & 15;
  const int halfB = (lane >> 4) << 3;  // B operand: c offset 0 / 8
  const int halfA = (lane >> 4) << 3;  // A operand: c offset 0 / 8

  float minv[4][8];
  int   mini[4][8];
  const v8f zero8 = {0.f, 0.f, 0.f, 0.f, 0.f, 0.f, 0.f, 0.f};
#pragma unroll
  for (int t = 0; t < 4; ++t)
#pragma unroll
    for (int r = 0; r < 8; ++r) { minv[t][r] = 3.4e38f; mini[t][r] = 0; }

  for (int kt = wv; kt < KCODES / 16; kt += 4) {
    const int kcol = kt * 16 + lrow;  // this lane's codebook column
    const unsigned short* wh = w_hi + kcol * CODE_DIM + halfB;
    const unsigned short* wl = w_lo + kcol * CODE_DIM + halfB;
    v8f acc[4];
#pragma unroll
    for (int t = 0; t < 4; ++t) acc[t] = zero8;

    for (int c0 = 0; c0 < CODE_DIM; c0 += 32) {
      ABu bh, bl;
      bh.q[0] = *(const v4u*)(wh + c0);
      bh.q[1] = *(const v4u*)(wh + c0 + 16);
      bl.q[0] = *(const v4u*)(wl + c0);
      bl.q[1] = *(const v4u*)(wl + c0 + 16);
#pragma unroll
      for (int t = 0; t < 4; ++t) {
        int row = t * 16 + lrow;
        int cA  = c0 + halfA;
        int o0  = row * CODE_DIM + (((cA >> 3) ^ (row & 15)) << 3);
        int o1  = row * CODE_DIM + ((((cA + 16) >> 3) ^ (row & 15)) << 3);
        ABu ah, al;
        ah.q[0] = *(const v4u*)&lds_hi[o0];
        ah.q[1] = *(const v4u*)&lds_hi[o1];
        al.q[0] = *(const v4u*)&lds_lo[o0];
        al.q[1] = *(const v4u*)&lds_lo[o1];
        // fp32-class accuracy: hi*hi + lo*hi + hi*lo
        acc[t] = __builtin_amdgcn_wmma_f32_16x16x32_bf16(
            false, ah.v, false, bh.v, (short)0, acc[t], false, false);
        acc[t] = __builtin_amdgcn_wmma_f32_16x16x32_bf16(
            false, al.v, false, bh.v, (short)0, acc[t], false, false);
        acc[t] = __builtin_amdgcn_wmma_f32_16x16x32_bf16(
            false, ah.v, false, bl.v, (short)0, acc[t], false, false);
      }
    }

    float wn = wnorm[kcol];
#pragma unroll
    for (int t = 0; t < 4; ++t)
#pragma unroll
      for (int r = 0; r < 8; ++r) {
        float d = wn - 2.f * acc[t][r];   // + ||z||^2 is row-constant
        if (d < minv[t][r]) { minv[t][r] = d; mini[t][r] = kcol; }
      }
  }

  // Reduce across the 16 lanes of each half (C layout: lanes 0-15 = rows r,
  // lanes 16-31 = rows 8+r). Tie-break on smaller index (argmin semantics).
#pragma unroll
  for (int m = 1; m <= 8; m <<= 1) {
#pragma unroll
    for (int t = 0; t < 4; ++t)
#pragma unroll
      for (int r = 0; r < 8; ++r) {
        float ov = __shfl_xor(minv[t][r], m, 32);
        int   oi = __shfl_xor(mini[t][r], m, 32);
        if (ov < minv[t][r] || (ov == minv[t][r] && oi < mini[t][r])) {
          minv[t][r] = ov; mini[t][r] = oi;
        }
      }
  }

  __syncthreads();  // all waves done reading A tiles; reuse LDS
  float* red_v = (float*)lds_hi;
  int*   red_i = (int*)lds_lo;
  if (lrow == 0) {
    int rb = (lane >> 4) * 8;
#pragma unroll
    for (int t = 0; t < 4; ++t)
#pragma unroll
      for (int r = 0; r < 8; ++r) {
        int row = t * 16 + rb + r;
        red_v[wv * M_TILE + row] = minv[t][r];
        red_i[wv * M_TILE + row] = mini[t][r];
      }
  }
  __syncthreads();
  if (tid < M_TILE) {
    float bv = red_v[tid];
    int   bi = red_i[tid];
#pragma unroll
    for (int w = 1; w < 4; ++w) {
      float v  = red_v[w * M_TILE + tid];
      int   i2 = red_i[w * M_TILE + tid];
      if (v < bv || (v == bv && i2 < bi)) { bv = v; bi = i2; }
    }
    idx_out[n0 + tid] = bi;
  }
}

// ---------------------------------------------------------------------------
// Gather: out[b,c,h,w] = weight[idx[b,h,w], c], written twice (code and
// detached_code are numerically identical), plus idx cast to float.
// Writes fully coalesced; weight reads hit the 1 MB L2-resident table.
// ---------------------------------------------------------------------------
__global__ void gather_kernel(const float* __restrict__ w,
                              const int* __restrict__ idx,
                              float* __restrict__ out) {
  int o  = blockIdx.x * 256 + threadIdx.x;
  int hw = o & (HW - 1);
  int c  = (o >> 10) & (CODE_DIM - 1);
  int b  = o >> 18;
  int n  = (b << 10) | hw;
  int k  = idx[n];
  float v = w[k * CODE_DIM + c];
  out[o] = v;
  out[o + OUT_ELEMS] = v;
  if (o < NROWS) out[2 * OUT_ELEMS + o] = (float)idx[o];
}

extern "C" void kernel_launch(void* const* d_in, const int* in_sizes, int n_in,
                              void* d_out, int out_size, void* d_ws, size_t ws_size,
                              hipStream_t stream) {
  const float* z = (const float*)d_in[0];   // [32,256,32,32] f32
  const float* w = (const float*)d_in[1];   // [1024,256] f32
  float* out = (float*)d_out;

  // Workspace layout: w_hi (512KB) | w_lo (512KB) | wnorm (4KB) | idx (128KB)
  unsigned short* w_hi = (unsigned short*)d_ws;
  unsigned short* w_lo = w_hi + KCODES * CODE_DIM;
  float* wnorm = (float*)(w_lo + KCODES * CODE_DIM);
  int* idx = (int*)(wnorm + KCODES);

  prep_weight_kernel<<<KCODES / 8, 256, 0, stream>>>(w, w_hi, w_lo, wnorm);
  argmin_kernel<<<NROWS / M_TILE, THREADS, 0, stream>>>(z, w_hi, w_lo, wnorm, idx);
  gather_kernel<<<OUT_ELEMS / 256, 256, 0, stream>>>(w, idx, out);
}
